// DecoderGRU_82394652606980
// MI455X (gfx1250) — matrix operand order repaired and checked by reading, not compiled, not run on hardware
//
#include <hip/hip_runtime.h>
#include <hip/hip_bf16.h>

// Problem constants (match reference)
#define BATCH  2048
#define LAT    256      // latent L
#define HID    1024     // hidden H
#define DW     256      // decoder width D
#define NSTEPS 64
#define KX1    512      // D + L (GRU1 input width)
#define MT     32       // batch rows per workgroup (two WMMA M-fragments)
#define THREADS 256     // 8 wave32s
#define NWAVE  8

#define AS1 __attribute__((address_space(1)))   // global
#define AS3 __attribute__((address_space(3)))   // LDS
#define DEVINL static __device__ __attribute__((always_inline)) inline

typedef __attribute__((ext_vector_type(16))) __bf16   v16bf;
typedef __attribute__((ext_vector_type(8)))  float    v8f;
typedef __attribute__((ext_vector_type(4)))  unsigned u32x4;

union Frag { u32x4 q[2]; v16bf v; };

// ---- WMMA fragment loaders (bf16, 16x16x32) -------------------------------
// A matrix 16x32 (MxK) from row-major LDS, row stride in elements.
// lanes 0-15: row M=lane, K[k0..k0+7] (v0-3), K[k0+16..k0+23] (v4-7);
// lanes 16-31: K[k0+8..k0+15], K[k0+24..k0+31].
DEVINL v16bf load_a(const AS3 __bf16* base, int strideElems, int k0, int lane) {
    const AS3 char* p = (const AS3 char*)base
                      + (lane & 15) * strideElems * 2
                      + k0 * 2 + ((lane >> 4) << 4);
    Frag f;
    f.q[0] = *(const AS3 u32x4*)(p);
    f.q[1] = *(const AS3 u32x4*)(p + 32);
    return f.v;
}

// B matrix 32x16 (KxN) from row-major global weights W[N,K] (x @ W^T: lane n = row n).
// lanes 0-15: K[k0..k0+7], K[k0+16..k0+23] of column N=lane; lanes 16-31: K[k0+8..k0+15], K[k0+24..k0+31].
DEVINL v16bf load_b(const AS1 __bf16* wrow0, int K, int k0, int lane) {
    const AS1 char* p = (const AS1 char*)wrow0
                      + (lane & 15) * K * 2
                      + k0 * 2 + ((lane >> 4) << 4);
    Frag f;
    f.q[0] = *(const AS1 u32x4*)(p);
    f.q[1] = *(const AS1 u32x4*)(p + 32);
    return f.v;
}

DEVINL v8f wmma_bf16(v16bf a, v16bf b, v8f c) {
    return __builtin_amdgcn_wmma_f32_16x16x32_bf16(false, a, false, b, (short)0, c, false, false);
}

// Branch-free transcendentals on the v_exp_f32 / v_rcp_f32 TRANS pipes.
DEVINL float sigm(float x)      { return __builtin_amdgcn_rcpf(1.f + __expf(-x)); }
DEVINL float fast_tanh(float x) { return fmaf(-2.f, __builtin_amdgcn_rcpf(1.f + __expf(2.f * x)), 1.f); }

// One GRU cell layer for this workgroup's 32 batch rows.
// Each wave owns 8 column tiles (128 hidden units); per tile it accumulates
// r/z/n for BOTH 16-row M-fragments so each weight (B) fragment feeds 2 WMMAs.
// First k-iteration is peeled with a literal-zero C so the backend uses the
// WMMA inline 0 accumulator instead of mov-zero chains.
DEVINL void gru_layer(const AS3 __bf16* xL, int xStride, int Kx, const AS1 __bf16* Wih,
                      const AS3 __bf16* hL, const AS1 __bf16* Whh,
                      const AS1 float* bih, const AS1 float* bhh,
                      AS3 __bf16* hOut, int wave, int lane) {
    for (int i = 0; i < 8; ++i) {
        const int cb = (wave * 8 + i) * 16;
        const AS1 __bf16* wr = Wih + (cb          ) * Kx;
        const AS1 __bf16* wz = Wih + (cb +     HID) * Kx;
        const AS1 __bf16* wn = Wih + (cb + 2 * HID) * Kx;
        v8f ir0, iz0, in0, ir1, iz1, in1;
        {
            const v8f zc = {};
            v16bf a0 = load_a(xL,                xStride, 0, lane);
            v16bf a1 = load_a(xL + 16 * xStride, xStride, 0, lane);
            v16bf b;
            b = load_b(wr, Kx, 0, lane); ir0 = wmma_bf16(a0, b, zc); ir1 = wmma_bf16(a1, b, zc);
            b = load_b(wz, Kx, 0, lane); iz0 = wmma_bf16(a0, b, zc); iz1 = wmma_bf16(a1, b, zc);
            b = load_b(wn, Kx, 0, lane); in0 = wmma_bf16(a0, b, zc); in1 = wmma_bf16(a1, b, zc);
        }
        for (int k0 = 32; k0 < Kx; k0 += 32) {
            v16bf a0 = load_a(xL,                xStride, k0, lane);
            v16bf a1 = load_a(xL + 16 * xStride, xStride, k0, lane);
            v16bf b;
            b = load_b(wr, Kx, k0, lane); ir0 = wmma_bf16(a0, b, ir0); ir1 = wmma_bf16(a1, b, ir1);
            b = load_b(wz, Kx, k0, lane); iz0 = wmma_bf16(a0, b, iz0); iz1 = wmma_bf16(a1, b, iz1);
            b = load_b(wn, Kx, k0, lane); in0 = wmma_bf16(a0, b, in0); in1 = wmma_bf16(a1, b, in1);
        }
        const AS1 __bf16* vr = Whh + (cb          ) * HID;
        const AS1 __bf16* vz = Whh + (cb +     HID) * HID;
        const AS1 __bf16* vn = Whh + (cb + 2 * HID) * HID;
        v8f hr0, hz0, hn0, hr1, hz1, hn1;
        {
            const v8f zc = {};
            v16bf a0 = load_a(hL,            HID, 0, lane);
            v16bf a1 = load_a(hL + 16 * HID, HID, 0, lane);
            v16bf b;
            b = load_b(vr, HID, 0, lane); hr0 = wmma_bf16(a0, b, zc); hr1 = wmma_bf16(a1, b, zc);
            b = load_b(vz, HID, 0, lane); hz0 = wmma_bf16(a0, b, zc); hz1 = wmma_bf16(a1, b, zc);
            b = load_b(vn, HID, 0, lane); hn0 = wmma_bf16(a0, b, zc); hn1 = wmma_bf16(a1, b, zc);
        }
        for (int k0 = 32; k0 < HID; k0 += 32) {
            v16bf a0 = load_a(hL,            HID, k0, lane);
            v16bf a1 = load_a(hL + 16 * HID, HID, k0, lane);
            v16bf b;
            b = load_b(vr, HID, k0, lane); hr0 = wmma_bf16(a0, b, hr0); hr1 = wmma_bf16(a1, b, hr1);
            b = load_b(vz, HID, k0, lane); hz0 = wmma_bf16(a0, b, hz0); hz1 = wmma_bf16(a1, b, hz1);
            b = load_b(vn, HID, k0, lane); hn0 = wmma_bf16(a0, b, hn0); hn1 = wmma_bf16(a1, b, hn1);
        }
        // Elementwise GRU update on the D fragments:
        // lanes 0-15 -> rows 0-7 (vgpr j), lanes 16-31 -> rows 8-15; +16 for fragment 1.
        const int col  = cb + (lane & 15);
        const int mrow = (lane >> 4) << 3;
        const float br_ = bih[col]           + bhh[col];
        const float bz_ = bih[col + HID]     + bhh[col + HID];
        const float bin = bih[col + 2 * HID];
        const float bhn = bhh[col + 2 * HID];
        #pragma unroll
        for (int j = 0; j < 8; ++j) {
            const int m = mrow + j;
            {   // fragment 0: rows 0..15
                float r  = sigm(ir0[j] + hr0[j] + br_);
                float zg = sigm(iz0[j] + hz0[j] + bz_);
                float n  = fast_tanh(in0[j] + bin + r * (hn0[j] + bhn));
                float hp = (float)hL[m * HID + col];
                hOut[m * HID + col] = (__bf16)((1.f - zg) * n + zg * hp);
            }
            {   // fragment 1: rows 16..31
                float r  = sigm(ir1[j] + hr1[j] + br_);
                float zg = sigm(iz1[j] + hz1[j] + bz_);
                float n  = fast_tanh(in1[j] + bin + r * (hn1[j] + bhn));
                float hp = (float)hL[(m + 16) * HID + col];
                hOut[(m + 16) * HID + col] = (__bf16)((1.f - zg) * n + zg * hp);
            }
        }
    }
}

extern "C" __global__ void __launch_bounds__(THREADS)
decoder_gru_persistent(const float* __restrict__ z_,
                       const float* __restrict__ b_init_,
                       const float* __restrict__ bih1_, const float* __restrict__ bhh1_,
                       const float* __restrict__ bih2_, const float* __restrict__ bhh2_,
                       const float* __restrict__ b_out_,
                       const __bf16* __restrict__ Wi_,
                       const __bf16* __restrict__ Wih1_, const __bf16* __restrict__ Whh1_,
                       const __bf16* __restrict__ Wih2_, const __bf16* __restrict__ Whh2_,
                       const __bf16* __restrict__ Wout_,
                       float* __restrict__ out_) {
    // Pin address spaces once: AS1 = global, AS3 = LDS.
    const AS1 float*  z      = (const AS1 float*)z_;
    const AS1 float*  b_init = (const AS1 float*)b_init_;
    const AS1 float*  bih1   = (const AS1 float*)bih1_;
    const AS1 float*  bhh1   = (const AS1 float*)bhh1_;
    const AS1 float*  bih2   = (const AS1 float*)bih2_;
    const AS1 float*  bhh2   = (const AS1 float*)bhh2_;
    const AS1 float*  b_out  = (const AS1 float*)b_out_;
    const AS1 __bf16* Wi     = (const AS1 __bf16*)Wi_;
    const AS1 __bf16* Wih1   = (const AS1 __bf16*)Wih1_;
    const AS1 __bf16* Whh1   = (const AS1 __bf16*)Whh1_;
    const AS1 __bf16* Wih2   = (const AS1 __bf16*)Wih2_;
    const AS1 __bf16* Whh2   = (const AS1 __bf16*)Whh2_;
    const AS1 __bf16* Wout   = (const AS1 __bf16*)Wout_;
    AS1 float*        out    = (AS1 float*)out_;

    extern __shared__ char smem_raw[];
    AS3 char*   smem = (AS3 char*)smem_raw;
    AS3 __bf16* XA = (AS3 __bf16*)smem;                                    // [32][512]  (out | z)
    AS3 __bf16* H1 = (AS3 __bf16*)(smem + MT * KX1 * 2);                   // 2 x [32][1024]
    AS3 __bf16* H2 = (AS3 __bf16*)(smem + MT * KX1 * 2 + 2 * MT * HID * 2);

    const int tid  = threadIdx.x;
    const int lane = tid & 31;
    const int wave = tid >> 5;
    const int row0 = blockIdx.x * MT;

    // Stage out0 (one-hot at col 255) and z (bf16) into the input slab.
    for (int idx = tid; idx < MT * 256; idx += THREADS) {
        const int r = idx >> 8, c = idx & 255;
        XA[r * KX1 + c]       = (__bf16)((c == 255) ? 1.f : 0.f);
        XA[r * KX1 + 256 + c] = (__bf16)z[(row0 + r) * LAT + c];
    }
    __syncthreads();

    // h1_0 = tanh(z @ W_init^T + b_init)  -> H1 buffer 0
    for (int i = 0; i < 8; ++i) {
        const int cb = (wave * 8 + i) * 16;
        const AS1 __bf16* w = Wi + cb * LAT;
        v8f acc0, acc1;
        {
            const v8f zc = {};
            v16bf a0 = load_a(XA + 256,            KX1, 0, lane);
            v16bf a1 = load_a(XA + 256 + 16 * KX1, KX1, 0, lane);
            v16bf b  = load_b(w, LAT, 0, lane);
            acc0 = wmma_bf16(a0, b, zc);
            acc1 = wmma_bf16(a1, b, zc);
        }
        for (int k0 = 32; k0 < LAT; k0 += 32) {
            v16bf a0 = load_a(XA + 256,            KX1, k0, lane);
            v16bf a1 = load_a(XA + 256 + 16 * KX1, KX1, k0, lane);
            v16bf b  = load_b(w, LAT, k0, lane);
            acc0 = wmma_bf16(a0, b, acc0);
            acc1 = wmma_bf16(a1, b, acc1);
        }
        const int col  = cb + (lane & 15);
        const int mrow = (lane >> 4) << 3;
        const float bi = b_init[col];
        #pragma unroll
        for (int j = 0; j < 8; ++j) {
            H1[(mrow + j)      * HID + col] = (__bf16)fast_tanh(acc0[j] + bi);
            H1[(mrow + j + 16) * HID + col] = (__bf16)fast_tanh(acc1[j] + bi);
        }
    }
    __syncthreads();

    for (int s = 0; s < NSTEPS; ++s) {
        AS3 __bf16* h1p = H1 + (s & 1) * MT * HID;
        AS3 __bf16* h1n = H1 + ((s + 1) & 1) * MT * HID;
        gru_layer(XA, KX1, KX1, Wih1, h1p, Whh1, bih1, bhh1, h1n, wave, lane);
        __syncthreads();

        AS3 __bf16* h2p = (s == 0) ? h1n : (H2 + ((s + 1) & 1) * MT * HID);  // h2 := h1 at step 0
        AS3 __bf16* h2n = H2 + (s & 1) * MT * HID;
        gru_layer(h1n, HID, HID, Wih2, h2p, Whh2, bih2, bhh2, h2n, wave, lane);
        __syncthreads();

        // Output: tmp = h2 @ W_out^T + b_out; log_softmax over class pair (c, c+128);
        // argmax one-hot feedback in interleaved [2c, 2c+1] layout.
        {
            const int c0 = wave * 16 + (lane & 15);           // 0..127 (pitch index)
            const AS1 __bf16* w0 = Wout + (wave * 16) * HID;
            const AS1 __bf16* w1 = Wout + (128 + wave * 16) * HID;
            v8f a00, a01, a10, a11;   // class{0,1} x frag{0,1}
            {
                const v8f zc = {};
                v16bf a0 = load_a(h2n,            HID, 0, lane);
                v16bf a1 = load_a(h2n + 16 * HID, HID, 0, lane);
                v16bf b;
                b = load_b(w0, HID, 0, lane); a00 = wmma_bf16(a0, b, zc); a01 = wmma_bf16(a1, b, zc);
                b = load_b(w1, HID, 0, lane); a10 = wmma_bf16(a0, b, zc); a11 = wmma_bf16(a1, b, zc);
            }
            for (int k0 = 32; k0 < HID; k0 += 32) {
                v16bf a0 = load_a(h2n,            HID, k0, lane);
                v16bf a1 = load_a(h2n + 16 * HID, HID, k0, lane);
                v16bf b;
                b = load_b(w0, HID, k0, lane); a00 = wmma_bf16(a0, b, a00); a01 = wmma_bf16(a1, b, a01);
                b = load_b(w1, HID, k0, lane); a10 = wmma_bf16(a0, b, a10); a11 = wmma_bf16(a1, b, a11);
            }
            const float bo0 = b_out[c0], bo1 = b_out[128 + c0];
            const int mrow = (lane >> 4) << 3;
            #pragma unroll
            for (int j = 0; j < 8; ++j) {
                #pragma unroll
                for (int f = 0; f < 2; ++f) {
                    const int m = mrow + j + 16 * f;
                    const float va = (f ? a01[j] : a00[j]) + bo0;
                    const float vb = (f ? a11[j] : a10[j]) + bo1;
                    const float mx = fmaxf(va, vb);
                    const float lse = mx + __logf(__expf(va - mx) + __expf(vb - mx));
                    const size_t base = ((size_t)(row0 + m) * NSTEPS + s) * DW;
                    out[base + c0]       = va - lse;
                    out[base + 128 + c0] = vb - lse;
                    // argmax over class (ties -> class 0), one-hot interleaved
                    const unsigned pack = (va >= vb) ? 0x00003F80u : 0x3F800000u;
                    *(AS3 unsigned*)(XA + m * KX1 + 2 * c0) = pack;
                }
            }
        }
        __syncthreads();
    }
}

// fp32 -> bf16 weight conversion (weights then live in L2 for the whole run)
extern "C" __global__ void cvt_f32_bf16(const float* __restrict__ src,
                                        __bf16* __restrict__ dst, int n) {
    int i = blockIdx.x * blockDim.x + threadIdx.x;
    if (i < n) dst[i] = (__bf16)src[i];
}

extern "C" void kernel_launch(void* const* d_in, const int* in_sizes, int n_in,
                              void* d_out, int out_size, void* d_ws, size_t ws_size,
                              hipStream_t stream) {
    const float* z     = (const float*)d_in[0];
    const float* Wi_f  = (const float*)d_in[1];
    const float* b_i   = (const float*)d_in[2];
    const float* Wih1f = (const float*)d_in[3];
    const float* Whh1f = (const float*)d_in[4];
    const float* bih1  = (const float*)d_in[5];
    const float* bhh1  = (const float*)d_in[6];
    const float* Wih2f = (const float*)d_in[7];
    const float* Whh2f = (const float*)d_in[8];
    const float* bih2  = (const float*)d_in[9];
    const float* bhh2  = (const float*)d_in[10];
    const float* Woutf = (const float*)d_in[11];
    const float* b_o   = (const float*)d_in[12];
    float* out = (float*)d_out;

    // bf16 weight arena in workspace (~22 MB; resident in the 192 MB L2)
    __bf16* ws = (__bf16*)d_ws;
    const size_t nWi   = (size_t)HID * LAT;       // 262144
    const size_t nWih1 = (size_t)3 * HID * KX1;   // 1572864
    const size_t nWhh  = (size_t)3 * HID * HID;   // 3145728
    const size_t nWout = (size_t)DW * HID;        // 262144
    __bf16* Wi   = ws;
    __bf16* Wih1 = Wi   + nWi;
    __bf16* Whh1 = Wih1 + nWih1;
    __bf16* Wih2 = Whh1 + nWhh;
    __bf16* Whh2 = Wih2 + nWhh;
    __bf16* Wout = Whh2 + nWhh;

    auto cvt = [&](const float* s, __bf16* d, size_t n) {
        cvt_f32_bf16<<<(unsigned)((n + 255) / 256), 256, 0, stream>>>(s, d, (int)n);
    };
    cvt(Wi_f,  Wi,   nWi);
    cvt(Wih1f, Wih1, nWih1);
    cvt(Whh1f, Whh1, nWhh);
    cvt(Wih2f, Wih2, nWhh);
    cvt(Whh2f, Whh2, nWhh);
    cvt(Woutf, Wout, nWout);

    const size_t smemBytes = (size_t)MT * KX1 * 2 + 4 * (size_t)MT * HID * 2; // 294912
    (void)hipFuncSetAttribute((const void*)decoder_gru_persistent,
                              hipFuncAttributeMaxDynamicSharedMemorySize,
                              (int)smemBytes);
    decoder_gru_persistent<<<BATCH / MT, THREADS, smemBytes, stream>>>(
        z, b_i, bih1, bhh1, bih2, bhh2, b_o,
        Wi, Wih1, Whh1, Wih2, Whh2, Wout, out);
}
